// CrossAttention2d_11330123726950
// MI455X (gfx1250) — matrix operand order repaired and checked by reading, not compiled, not run on hardware
//
#include <hip/hip_runtime.h>

typedef __attribute__((ext_vector_type(16))) _Float16 v16h;
typedef __attribute__((ext_vector_type(8)))  _Float16 v8h;
typedef __attribute__((ext_vector_type(8)))  float    v8f;

#define HW   4096
#define NTOK 1024
#define DH   64
#define CCH  256
#define DVIT 768

__device__ __forceinline__ v8f wmma32f16(v16h a, v16h b, v8f c) {
  return __builtin_amdgcn_wmma_f32_16x16x32_f16(false, a, false, b, (short)0, c, false, false);
}

// ---- fragment loaders -------------------------------------------------------
// A operand, 16x32 f16 tile. Lane's row = lane&15. Element e<8 -> K = k0+8*hs+e,
// e>=8 -> K = k0+16+8*hs+(e-8). Two contiguous 8-half (16B) runs.
__device__ __forceinline__ v16h load_a_h16(const _Float16* __restrict__ rowbase, int hs, int k0) {
  v8h lo = *(const v8h*)(rowbase + k0 + 8 * hs);
  v8h hi = *(const v8h*)(rowbase + k0 + 16 + 8 * hs);
  v16h r;
#pragma unroll
  for (int i = 0; i < 8; ++i) { r[i] = lo[i]; r[i + 8] = hi[i]; }
  return r;
}

// B operand, 32x16 f16 tile. Lane's col = lane&15. Element e<8 -> K = k0+8*hs+e,
// e>=8 -> K = k0+16+8*hs+(e-8). Two contiguous 8-half (16B) runs.
__device__ __forceinline__ v16h load_b_h16(const _Float16* __restrict__ colbase, int hs, int k0) {
  v8h lo = *(const v8h*)(colbase + k0 + 8 * hs);
  v8h hi = *(const v8h*)(colbase + k0 + 16 + 8 * hs);
  v16h r;
#pragma unroll
  for (int i = 0; i < 8; ++i) { r[i] = lo[i]; r[i + 8] = hi[i]; }
  return r;
}

// A operand from fp32 source with compile-time element stride (1 = contiguous,
// 4096 = channel gather out of BCHW conv features), converted to f16.
template <int STRIDE>
__device__ __forceinline__ v16h load_a_f32(const float* __restrict__ rowbase, int hs, int k0) {
  v16h r;
  const int klo = k0 + 8 * hs;
  const int khi = klo + 16;
#pragma unroll
  for (int i = 0; i < 8; ++i) {
    r[i]     = (_Float16)rowbase[(size_t)(klo + i) * STRIDE];
    r[i + 8] = (_Float16)rowbase[(size_t)(khi + i) * STRIDE];
  }
  return r;
}

// B operand from fp32 weight row (contiguous over K), converted to f16.
__device__ __forceinline__ v16h load_b_f32(const float* __restrict__ colbase, int hs, int k0) {
  v16h r;
  const int k = k0 + 8 * hs;
#pragma unroll
  for (int i = 0; i < 8; ++i) { r[i] = (_Float16)colbase[k + i]; r[i + 8] = (_Float16)colbase[k + 16 + i]; }
  return r;
}

// ---- xor-16 lane exchange via v_permlanex16 (VALU, no LDS) ------------------
__device__ __forceinline__ float xor16_f32(float v) {
  const int i = __float_as_int(v);
  const int r = __builtin_amdgcn_permlanex16(i, i, 0x76543210, (int)0xFEDCBA98u, false, false);
  return __int_as_float(r);
}

// ============================================================================
// GEMM + bias -> f16, out = (A[rows x K] * W[64 x K]^T + bias) * oscale.
// One wave computes a 16x64 tile. Block = 4 waves = 4 row tiles.
// ============================================================================
template <int ACS>
__global__ void proj_kernel(const float* __restrict__ A, long a_bstride, int a_rstride,
                            const float* __restrict__ W, const float* __restrict__ bias, int Kdim,
                            _Float16* __restrict__ out, long o_bstride, int o_rstride, int o_cstride,
                            float oscale) {
  const int lane = threadIdx.x & 31;
  const int wv   = threadIdx.x >> 5;
  const int hs   = lane >> 4;
  const int ln   = lane & 15;
  const int batch = blockIdx.y;
  const int row0  = (blockIdx.x * 4 + wv) * 16;

  const float* arow = A + (size_t)batch * a_bstride + (size_t)(row0 + ln) * a_rstride;

  v8f acc[4] = {};
  for (int k0 = 0; k0 < Kdim; k0 += 32) {
    v16h a = load_a_f32<ACS>(arow, hs, k0);
#pragma unroll
    for (int t = 0; t < 4; ++t) {
      v16h bf = load_b_f32(W + (size_t)(t * 16 + ln) * Kdim, hs, k0);
      acc[t] = wmma32f16(a, bf, acc[t]);
    }
  }

  _Float16* obase = out + (size_t)batch * o_bstride;
#pragma unroll
  for (int t = 0; t < 4; ++t) {
    const int col = t * 16 + ln;
    const float bv = bias[col];
#pragma unroll
    for (int r = 0; r < 8; ++r) {
      const int row = row0 + r + 8 * hs;
      obase[(size_t)row * o_rstride + (size_t)col * o_cstride] =
          (_Float16)((acc[t][r] + bv) * oscale);
    }
  }
}

// ============================================================================
// Flash attention, transposed-score formulation.
// Each wave: 16 queries (one per lane column) x full d=64, keys streamed in
// chunks of 32.  S^T = K @ Q^T  (A = K tile, B = Q tile, loop-invariant).
// In D layout each lane owns one query column with 16 of its 32 key-scores in
// registers; the other 16 live in the xor-16 paired lane -> softmax reductions
// are in-lane trees + one v_permlanex16.  m/l/corr are one scalar per lane.
// O^T = V^T @ P^T accumulated via WMMA; P^T staged through LDS as [q][n] so
// both store and load sides are packed b128 accesses.
// qh[4096][64] is pre-scaled by 0.125*log2(e); softmax uses exp2.
// ============================================================================
__global__ void attn_kernel(const _Float16* __restrict__ qh, const _Float16* __restrict__ kh,
                            const _Float16* __restrict__ vhT, _Float16* __restrict__ oh) {
  __shared__ alignas(16) _Float16 pstage[4][16 * 32];   // per-wave P^T stage, [q][n]

  const int lane = threadIdx.x & 31;
  const int wv   = threadIdx.x >> 5;
  const int hs   = lane >> 4;
  const int ln   = lane & 15;
  const int batch = blockIdx.y;
  const int q0    = (blockIdx.x * 4 + wv) * 16;

  const _Float16* qrow  = qh  + (size_t)batch * HW * DH + (size_t)(q0 + ln) * DH;
  const _Float16* kbase = kh  + (size_t)batch * NTOK * DH;
  const _Float16* vbase = vhT + (size_t)batch * DH * NTOK;

  // Q as loop-invariant B operands [32dh x 16q]; lane col = query q0+ln.
  const v16h qb0 = load_b_h16(qrow, hs, 0);
  const v16h qb1 = load_b_h16(qrow, hs, 32);

  float m = -3.0e38f, l = 0.0f;
  v8f acc[4] = {};                                      // O^T: row d = r+8*hs+16*t, col q = ln
  _Float16* pst = &pstage[wv][0];

  for (int n0 = 0; n0 < NTOK; n0 += 32) {
    // ---- S^T tiles [32 keys x 16 q]; rows = keys, cols = queries ----
    v8f T0 = {}, T1 = {};
    {
      const _Float16* kr0 = kbase + (size_t)(n0 + ln) * DH;
      T0 = wmma32f16(load_a_h16(kr0, hs, 0),  qb0, T0);
      T0 = wmma32f16(load_a_h16(kr0, hs, 32), qb1, T0);
      const _Float16* kr1 = kbase + (size_t)(n0 + 16 + ln) * DH;
      T1 = wmma32f16(load_a_h16(kr1, hs, 0),  qb0, T1);
      T1 = wmma32f16(load_a_h16(kr1, hs, 32), qb1, T1);
    }

    // ---- per-query chunk max: in-lane tree over 16 scores + paired lane ----
    float cm = T0[0];
#pragma unroll
    for (int r = 1; r < 8; ++r) cm = fmaxf(cm, T0[r]);
#pragma unroll
    for (int r = 0; r < 8; ++r) cm = fmaxf(cm, T1[r]);
    cm = fmaxf(cm, xor16_f32(cm));

    const float mnew = fmaxf(m, cm);
    const float corr = __builtin_amdgcn_exp2f(m - mnew);
    m = mnew;

    float p0[8], p1[8];
    float rs = 0.0f;
#pragma unroll
    for (int r = 0; r < 8; ++r) {
      p0[r] = __builtin_amdgcn_exp2f(T0[r] - mnew);
      p1[r] = __builtin_amdgcn_exp2f(T1[r] - mnew);
      rs += p0[r] + p1[r];
    }
    rs += xor16_f32(rs);
    l = l * corr + rs;

#pragma unroll
    for (int t = 0; t < 4; ++t)
#pragma unroll
      for (int r = 0; r < 8; ++r) acc[t][r] *= corr;

    // ---- stage P^T as [q][n]: packed 8-half vector stores, no scatter ----
    v8h ph0, ph1;
#pragma unroll
    for (int r = 0; r < 8; ++r) { ph0[r] = (_Float16)p0[r]; ph1[r] = (_Float16)p1[r]; }
    *(v8h*)(pst + ln * 32 + 8 * hs)      = ph0;         // keys n0 + r+8*hs
    *(v8h*)(pst + ln * 32 + 16 + 8 * hs) = ph1;         // keys n0+16 + r+8*hs
    asm volatile("s_wait_dscnt 0" ::: "memory");        // DS in-order; fence compiler + data
    const v16h pb = load_b_h16(pst + ln * 32, hs, 0);   // B operand [32n x 16q]
    asm volatile("s_wait_dscnt 0" ::: "memory");        // keep next iter's stores after loads

    // ---- O^T += V^T[64d x 32n] @ P^T[32n x 16q] ----
#pragma unroll
    for (int t = 0; t < 4; ++t) {
      const _Float16* vr = vbase + (size_t)(t * 16 + ln) * NTOK;
      acc[t] = wmma32f16(load_a_h16(vr, hs, n0), pb, acc[t]);
    }
  }

  // ---- epilogue: lane = query, elements = consecutive d -> packed stores ----
  const float inv = 1.0f / l;
  _Float16* orow = oh + (size_t)batch * HW * DH + (size_t)(q0 + ln) * DH;
#pragma unroll
  for (int t = 0; t < 4; ++t) {
    v8h oph;
#pragma unroll
    for (int r = 0; r < 8; ++r) oph[r] = (_Float16)(acc[t][r] * inv);
    *(v8h*)(orow + t * 16 + 8 * hs) = oph;              // d = 16*t + 8*hs + r
  }
}

// ============================================================================
// Output projection: out[b,c,hw] = sum_d o[b,hw,d]*out_w[c,d] + out_b[c].
// Block = 16 query rows x 256 channels (4 waves x 64-channel tiles).
// ============================================================================
__global__ void outproj_kernel(const _Float16* __restrict__ oh, const float* __restrict__ ow,
                               const float* __restrict__ ob, float* __restrict__ out) {
  const int lane = threadIdx.x & 31;
  const int wv   = threadIdx.x >> 5;
  const int hs   = lane >> 4;
  const int ln   = lane & 15;
  const int batch = blockIdx.y;
  const int q0    = blockIdx.x * 16;
  const int c0    = wv * 64;

  const _Float16* arow = oh + (size_t)batch * HW * DH + (size_t)(q0 + ln) * DH;
  const v16h a0 = load_a_h16(arow, hs, 0);
  const v16h a1 = load_a_h16(arow, hs, 32);

  float* obase = out + (size_t)batch * CCH * HW;
#pragma unroll
  for (int t = 0; t < 4; ++t) {
    const int c = c0 + t * 16 + ln;
    const float* wrow = ow + (size_t)c * DH;
    v8f d = {};
    d = wmma32f16(a0, load_b_f32(wrow, hs, 0),  d);
    d = wmma32f16(a1, load_b_f32(wrow, hs, 32), d);
    const float bv = ob[c];
#pragma unroll
    for (int r = 0; r < 8; ++r)
      obase[(size_t)c * HW + (q0 + r + 8 * hs)] = d[r] + bv;
  }
}

// ============================================================================
extern "C" void kernel_launch(void* const* d_in, const int* in_sizes, int n_in,
                              void* d_out, int out_size, void* d_ws, size_t ws_size,
                              hipStream_t stream) {
  (void)in_sizes; (void)n_in; (void)out_size; (void)ws_size;
  const float* conv = (const float*)d_in[0];
  const float* vit  = (const float*)d_in[1];
  const float* qw   = (const float*)d_in[2];
  const float* qb   = (const float*)d_in[3];
  const float* kw   = (const float*)d_in[4];
  const float* kb   = (const float*)d_in[5];
  const float* vw   = (const float*)d_in[6];
  const float* vb   = (const float*)d_in[7];
  const float* ow   = (const float*)d_in[8];
  const float* obv  = (const float*)d_in[9];
  float* out = (float*)d_out;

  char* ws = (char*)d_ws;
  _Float16* qh  = (_Float16*)(ws);                         // 8*4096*64 f16 = 4 MB
  _Float16* kh  = (_Float16*)(ws + 4u * 1024 * 1024);      // 8*1024*64 f16 = 1 MB
  _Float16* vhT = (_Float16*)(ws + 5u * 1024 * 1024);      // 8*64*1024 f16 = 1 MB
  _Float16* oh  = (_Float16*)(ws + 6u * 1024 * 1024);      // 8*4096*64 f16 = 4 MB

  const dim3 blk(128);
  // K projection: vit[1024x768] * kw^T -> kh[1024][64] (row-major)
  proj_kernel<1><<<dim3(16, 8), blk, 0, stream>>>(
      vit, (long)NTOK * DVIT, DVIT, kw, kb, DVIT, kh, (long)NTOK * DH, DH, 1, 1.0f);
  // V projection: -> vhT[64][1024] (transposed, contiguous over keys)
  proj_kernel<1><<<dim3(16, 8), blk, 0, stream>>>(
      vit, (long)NTOK * DVIT, DVIT, vw, vb, DVIT, vhT, (long)DH * NTOK, 1, NTOK, 1.0f);
  // Q projection: conv BCHW channel-gather; fold softmax scale and log2(e)
  // (exp2-based softmax): 0.125 * 1.4426950408889634
  proj_kernel<HW><<<dim3(64, 8), blk, 0, stream>>>(
      conv, (long)CCH * HW, 1, qw, qb, CCH, qh, (long)HW * DH, DH, 1, 0.18033688011112042f);
  // Flash attention (transposed-score)
  attn_kernel<<<dim3(64, 8), blk, 0, stream>>>(qh, kh, vhT, oh);
  // Output projection back to BCHW fp32
  outproj_kernel<<<dim3(256, 8), blk, 0, stream>>>(oh, ow, obv, out);
}
